// DynamicConv_33079838113867
// MI455X (gfx1250) — matrix operand order repaired and checked by reading, not compiled, not run on hardware
//
#include <hip/hip_runtime.h>

typedef __attribute__((ext_vector_type(16))) _Float16 v16h;
typedef __attribute__((ext_vector_type(2)))  __fp16   v2fp;
typedef __attribute__((ext_vector_type(8)))  float    v8f;

namespace {
constexpr int B_   = 16,  CIN = 64, COUT = 64, H_ = 256, W_ = 256;
constexpr int KS_  = 5,   HO  = 127, WO = 127, NTOT = HO * WO;   // 16129
constexpr int KTOT = CIN * KS_ * KS_;                            // 1600
constexpr int KSTEP = 32, BN = 128;                              // N tile per block
constexpr int NITER = KTOT / KSTEP;                              // 50
constexpr int NBLK = (NTOT + BN - 1) / BN;                       // 127
constexpr size_t OUT0 = (size_t)B_ * COUT * NTOT;                // 16,516,096
}

// LDS-only split barrier: make LDS stores visible and synchronize the workgroup
// WITHOUT draining in-flight global loads (unlike __syncthreads, which emits
// s_wait_loadcnt_dscnt 0 and kills the software-pipelined gathers).
__device__ __forceinline__ void lds_barrier() {
  asm volatile("s_wait_dscnt 0x0\n\t"
               "s_barrier_signal -1\n\t"
               "s_barrier_wait -1" ::: "memory");
}

// ---------------- 1) global average pool: x (16,64,256,256) -> pooled (16,64)
__global__ __launch_bounds__(256) void pool_kernel(const float* __restrict__ x,
                                                   float* __restrict__ pooled) {
  __shared__ float red[256];
  const int bc = blockIdx.x;                       // b*64 + c
  const float4* p4 = (const float4*)(x + (size_t)bc * (H_ * W_));
  float s = 0.f;
  for (int i = threadIdx.x; i < (H_ * W_) / 4; i += 256) {
    const float4 v = p4[i];
    s += (v.x + v.y) + (v.z + v.w);
  }
  red[threadIdx.x] = s;
  __syncthreads();
  for (int off = 128; off > 0; off >>= 1) {
    if ((int)threadIdx.x < off) red[threadIdx.x] += red[threadIdx.x + off];
    __syncthreads();
  }
  if (threadIdx.x == 0) pooled[bc] = red[0] * (1.f / (H_ * W_));
}

// ---------------- 2) attention MLP + softmax + agg_b
__global__ __launch_bounds__(256) void att_kernel(
    const float* __restrict__ pooled, const float* __restrict__ fc1w,
    const float* __restrict__ fc1b, const float* __restrict__ fc2w,
    const float* __restrict__ fc2b, const float* __restrict__ biasKC,
    float* __restrict__ att, float* __restrict__ aggb) {
  __shared__ float attS[B_ * 4];
  const int tid = threadIdx.x;
  if (tid < B_) {
    const int b = tid;
    float a[4];
#pragma unroll
    for (int k = 0; k < 4; ++k) {
      float s = fc1b[k];
      for (int c = 0; c < CIN; ++c) s += pooled[b * CIN + c] * fc1w[k * CIN + c];
      a[k] = fmaxf(s, 0.f);
    }
    float z[4], mx = -1e30f;
#pragma unroll
    for (int k = 0; k < 4; ++k) {
      float s = fc2b[k];
#pragma unroll
      for (int j = 0; j < 4; ++j) s += a[j] * fc2w[k * 4 + j];
      z[k] = s;
      mx = fmaxf(mx, s);
    }
    float den = 0.f;
#pragma unroll
    for (int k = 0; k < 4; ++k) { z[k] = __expf(z[k] - mx); den += z[k]; }
    const float inv = 1.f / den;
#pragma unroll
    for (int k = 0; k < 4; ++k) {
      const float v = z[k] * inv;
      att[b * 4 + k]  = v;
      attS[b * 4 + k] = v;
    }
  }
  __syncthreads();
  for (int i = tid; i < B_ * COUT; i += 256) {
    const int b = i >> 6, c = i & 63;
    float s = 0.f;
#pragma unroll
    for (int k = 0; k < 4; ++k) s += attS[b * 4 + k] * biasKC[k * COUT + c];
    aggb[i] = s;
  }
}

// ---------------- 3) agg_w in WMMA-fragment order [b][it][mtile][lane][j] (f16)
//                   + permuted f32 w_ret output. Linear idx == fragment order,
//                   so the f16 store is perfectly coalesced.
__global__ __launch_bounds__(256) void aggw_kernel(
    const float* __restrict__ weight, const float* __restrict__ att,
    _Float16* __restrict__ awh2, float* __restrict__ wret) {
  const int idx = blockIdx.x * 256 + threadIdx.x;
  if (idx >= B_ * KTOT * COUT) return;
  const int j     = idx & 15;
  const int lane  = (idx >> 4) & 31;
  const int mtile = (idx >> 9) & 3;
  const int itb   = idx >> 11;
  const int it    = itb % NITER;
  const int b     = itb / NITER;
  // inverse of A-fragment mapping (ISA 7.12.2, 16-bit A 16x32):
  const int kk = (j & 7) | ((lane & 16) ? 8 : 0) | ((j & 8) ? 16 : 0);
  const int k  = it * KSTEP + kk;
  const int co = mtile * 16 + (lane & 15);
  const int ci = k / 25;
  const int p  = k - ci * 25;
  float s = 0.f;
  const float* wp = weight + (size_t)co * KTOT + k;  // weight[(kb*64+co)*1600 + k]
#pragma unroll
  for (int kb = 0; kb < 4; ++kb)
    s += att[b * 4 + kb] * wp[(size_t)kb * COUT * KTOT];
  awh2[idx] = (_Float16)s;
  __builtin_nontemporal_store(
      s, &wret[(((size_t)(b * 64 + co)) * 25 + p) * 64 + ci]);
}

// ---------------- 4) per-sample implicit-GEMM conv via v_wmma_f32_16x16x32_f16
// Block 256 thr (8 waves). Tile M=64 x N=128, 50 K-steps of 32.
// A fragments straight from global (fragment-ordered, L2-resident).
// B tile staged to double-buffered LDS; one LDS-only barrier per step;
// gathers for step it+1 stay in flight across barrier + WMMA of step it.
__global__ __launch_bounds__(256) void conv_kernel(
    const float* __restrict__ x, const _Float16* __restrict__ awh2,
    const float* __restrict__ aggb, float* __restrict__ out) {
  constexpr int BBUF = 8 * 32 * 16;   // 4096 halves per buffer
  __shared__ __align__(32) _Float16 Bs[2 * BBUF];
  __shared__ int hwB[BN];             // packed (hi_base<<16)|(wi_base&0xffff)

  const int b    = blockIdx.y;
  const int n0   = blockIdx.x * BN;
  const int tid  = threadIdx.x;
  const int lane = tid & 31;
  const int wave = tid >> 5;
  const int wm   = wave & 3;
  const int wn   = wave >> 2;         // 0/1 -> ntiles wn*4..wn*4+3

  if (tid < BN) {
    const int ng = min(n0 + tid, NTOT - 1);        // clamp keeps ho,wo <= 126
    const int ho = ng / WO;
    const int wo = ng - ho * WO;
    hwB[tid] = ((ho * 2 - 1) << 16) | ((wo * 2 - 1) & 0xffff);
  }
  __syncthreads();

  const float*    xb  = x    + (size_t)b * CIN * H_ * W_;
  const _Float16* awb = awh2 + (size_t)b * (NITER * 4 * 32 * 16);

  // B staging: thread owns K-row pair (kk0,kk0+1) and 8 consecutive n.
  const int kp  = tid >> 4;                        // 0..15
  const int kk0 = kp * 2;
  const int e0B = (tid & 15) * 8;
  const int laneOffB = (kk0 & 8) ? 16 : 0;
  const int jB0 = (kk0 & 7) | ((kk0 & 16) ? 8 : 0); // even -> v2 store at (j,j+1)
  // halves; element i at base + i*16
  const int bStBase = (((e0B >> 4) * 32) + laneOffB + (e0B & 15)) * 16 + jB0;

  // Bounds facts: hi = hib+kh <= 255, wi = wib+kw <= 255 always; only <0 can trip.
  int offW[8];
  unsigned hneg = 0, wneg = 0, okN = 0;
#pragma unroll
  for (int i = 0; i < 8; ++i) {
    const int hw = hwB[e0B + i];
    const int hb = hw >> 16;
    const int wb = (hw << 16) >> 16;
    offW[i] = hb * W_ + wb;
    if (hb < 0) hneg |= 1u << i;
    if (wb < 0) wneg |= 1u << i;
    if (n0 + e0B + i < NTOT) okN |= 1u << i;
  }

  // incremental k -> (ci, p): k = kk0 + 32*it
  int ci = (kk0 >= 25) ? 1 : 0;
  int p  = kk0 - ci * 25;

  float    bv0[8], bv1[8];
  unsigned okm0 = 0, okm1 = 0;

  auto issue = [&]() {
    const int p1t = p + 1;
    const int c1  = ci + ((p1t == 25) ? 1 : 0);
    const int p1  = (p1t == 25) ? 0 : p1t;
    const int kh0 = (p * 205) >> 10,  kw0 = p - kh0 * 5;    // p/5, p%5
    const int kh1 = (p1 * 205) >> 10, kw1 = p1 - kh1 * 5;
    const int koff0 = (ci << 16) + (kh0 << 8) + kw0;        // ci*HW + kh*W + kw
    const int koff1 = (c1 << 16) + (kh1 << 8) + kw1;
    okm0 = okN & ~(((kh0 == 0) ? hneg : 0u) | ((kw0 == 0) ? wneg : 0u));
    okm1 = okN & ~(((kh1 == 0) ? hneg : 0u) | ((kw1 == 0) ? wneg : 0u));
#pragma unroll
    for (int i = 0; i < 8; ++i) {
      int t0 = offW[i] + koff0; t0 = (t0 < 0) ? 0 : t0;     // single v_max clamp
      int t1 = offW[i] + koff1; t1 = (t1 < 0) ? 0 : t1;
      bv0[i] = xb[(unsigned)t0];                            // scalar-base + voffset
      bv1[i] = xb[(unsigned)t1];
    }
  };

  issue();

  v8f acc0 = {}, acc1 = {}, acc2 = {}, acc3 = {};

  for (int it = 0; it < NITER; ++it) {
    _Float16* Bb = Bs + (it & 1) * BBUF;
    // drain staged registers into LDS as packed half2 (b32 stores)
#pragma unroll
    for (int i = 0; i < 8; ++i) {
      const int m0 = (int)(okm0 << (31 - i)) >> 31;
      const int m1 = (int)(okm1 << (31 - i)) >> 31;
      const float f0 = __int_as_float(__float_as_int(bv0[i]) & m0);
      const float f1 = __int_as_float(__float_as_int(bv1[i]) & m1);
      const v2fp hp = __builtin_amdgcn_cvt_pkrtz(f0, f1);   // 1 packed cvt/pair
      *(v2fp*)(Bb + bStBase + i * 16) = hp;
    }
    // pipeline: issue next slice's gathers now; they stay in flight across
    // the LDS-only barrier and the WMMA block below.
    if (it + 1 < NITER) {
      p += 32;
      if (p >= 25) { p -= 25; ++ci; }
      if (p >= 25) { p -= 25; ++ci; }
      issue();
    }
    // A fragment straight from global (fragment-ordered, L2-resident)
    const v16h af = *(const v16h*)(awb + (((it * 4) + wm) * 32 + lane) * 16);
    if (it + 1 < NITER)
      __builtin_prefetch(awb + (((it + 1) * 4 + wm) * 32 + lane) * 16, 0, 1);

    lds_barrier();     // s_wait_dscnt 0 + s_barrier_signal/wait: no loadcnt drain

    // interleave fragment loads with WMMA so waits are staged, not serialized
    const _Float16* Bw = Bb + ((wn * 4) * 32 + lane) * 16;
    v16h bf0 = *(const v16h*)(Bw);
    v16h bf1 = *(const v16h*)(Bw + 32 * 16);
    acc0 = __builtin_amdgcn_wmma_f32_16x16x32_f16(false, af, false, bf0,
                                                  (short)0, acc0, false, false);
    v16h bf2 = *(const v16h*)(Bw + 2 * 32 * 16);
    acc1 = __builtin_amdgcn_wmma_f32_16x16x32_f16(false, af, false, bf1,
                                                  (short)0, acc1, false, false);
    v16h bf3 = *(const v16h*)(Bw + 3 * 32 * 16);
    acc2 = __builtin_amdgcn_wmma_f32_16x16x32_f16(false, af, false, bf2,
                                                  (short)0, acc2, false, false);
    acc3 = __builtin_amdgcn_wmma_f32_16x16x32_f16(false, af, false, bf3,
                                                  (short)0, acc3, false, false);
  }

  // ---- epilogue: D layout (VGPR r: lanes0-15 M=r, lanes16-31 M=r+8; N=lane%16)
  const int coBase = wm * 16 + ((lane >= 16) ? 8 : 0);
  const int nr = lane & 15;
  float* outb = out + (size_t)b * COUT * NTOT;
#pragma unroll
  for (int r = 0; r < 8; ++r) {
    const int co = coBase + r;
    const float bb = aggb[b * COUT + co];
    const float v0 = acc0[r] + bb, v1 = acc1[r] + bb;
    const float v2 = acc2[r] + bb, v3 = acc3[r] + bb;
    const int ng0 = n0 + (wn * 4 + 0) * 16 + nr;
    const int ng1 = n0 + (wn * 4 + 1) * 16 + nr;
    const int ng2 = n0 + (wn * 4 + 2) * 16 + nr;
    const int ng3 = n0 + (wn * 4 + 3) * 16 + nr;
    float* row = outb + (size_t)co * NTOT;
    if (ng0 < NTOT) __builtin_nontemporal_store(v0, row + ng0);
    if (ng1 < NTOT) __builtin_nontemporal_store(v1, row + ng1);
    if (ng2 < NTOT) __builtin_nontemporal_store(v2, row + ng2);
    if (ng3 < NTOT) __builtin_nontemporal_store(v3, row + ng3);
  }
}

extern "C" void kernel_launch(void* const* d_in, const int* in_sizes, int n_in,
                              void* d_out, int out_size, void* d_ws, size_t ws_size,
                              hipStream_t stream) {
  const float* x      = (const float*)d_in[0];
  const float* fc1w   = (const float*)d_in[1];
  const float* fc1b   = (const float*)d_in[2];
  const float* fc2w   = (const float*)d_in[3];
  const float* fc2b   = (const float*)d_in[4];
  const float* weight = (const float*)d_in[5];
  const float* biasKC = (const float*)d_in[6];

  float* out  = (float*)d_out;
  float* wret = out + OUT0;

  char* ws = (char*)d_ws;
  float*    pooled = (float*)(ws);                  //  4 KB
  float*    att    = (float*)(ws + 4096);           //  256 B
  float*    aggb   = (float*)(ws + 8192);           //  4 KB
  _Float16* awh2   = (_Float16*)(ws + 16384);       //  3.125 MB (16,50,4,32,16) f16

  pool_kernel<<<dim3(B_ * CIN), dim3(256), 0, stream>>>(x, pooled);
  att_kernel<<<dim3(1), dim3(256), 0, stream>>>(pooled, fc1w, fc1b, fc2w, fc2b,
                                                biasKC, att, aggb);
  aggw_kernel<<<dim3((B_ * KTOT * COUT + 255) / 256), dim3(256), 0, stream>>>(
      weight, att, awh2, wret);
  conv_kernel<<<dim3(NBLK, B_), dim3(256), 0, stream>>>(x, awh2, aggb, out);
}
